// Causal_ML_79207786872998
// MI455X (gfx1250) — compile-verified, matrix operand order checked by reading, not run on hardware
//
#include <hip/hip_runtime.h>
#include <math.h>

// Problem constants (from reference): B=256, T=200, V=16, N=64, O=1
#define BB 256
#define TT 200
#define VV 16
#define NN 64
#define KIN   1040      // V + V*N
#define KT_GATE 33      // padded K=1056 -> 33 k-tiles of 32: [h(1024)|x(16)|pad(16)]
#define MTILE 32        // batch rows per block (2 WMMA row tiles per wave)

typedef __attribute__((ext_vector_type(16))) __bf16 v16bf;
typedef __attribute__((ext_vector_type(8)))  __bf16 v8bf;
typedef __attribute__((ext_vector_type(8)))  float  v8f;

union AFrag { v16bf v; v8bf h[2]; };

// ---------------------------------------------------------------------------
// Pack fp32 gate weights (out=1024, in=1040) into bf16 WMMA B-fragments.
// Fragment layout per (gate g, feature-tile ft[0..63], k-tile kt[0..32]):
//   512 halfs = 32 lanes x 16 halfs, lane holds column n = lane&15,
//   k_local = (e>>3)*16 + (lane>>4)*8 + (e&7)   (16-bit A/B VGPR map)
// K reorder: k' in [0,1024) -> original col 16+k' (h part),
//            k' in [1024,1040) -> original col k'-1024 (x part), else 0 pad.
// ---------------------------------------------------------------------------
__global__ void pack_gates_kernel(const float* __restrict__ Wi,
                                  const float* __restrict__ Wf,
                                  const float* __restrict__ Wo,
                                  __bf16* __restrict__ pw) {
  int i = blockIdx.x * blockDim.x + threadIdx.x;     // 3*64*33*512 total
  if (i >= 3 * 64 * KT_GATE * 512) return;
  int e   = i & 15;
  int l   = (i >> 4) & 31;
  int kt  = (i >> 9) % KT_GATE;
  int ftg = (i >> 9) / KT_GATE;                      // 0..191
  int g   = ftg / 64;
  int ft  = ftg % 64;
  int s = l >> 4, chunk = e >> 3, w = e & 7;
  int kl = chunk * 16 + s * 8 + w;
  int kg = kt * 32 + kl;
  int row = ft * 16 + (l & 15);                      // output feature 0..1023
  const float* Wg = (g == 0) ? Wi : ((g == 1) ? Wf : Wo);
  float val = 0.f;
  if (kg < 1024)      val = Wg[(size_t)row * KIN + (16 + kg)];
  else if (kg < 1040) val = Wg[(size_t)row * KIN + (kg - 1024)];
  pw[i] = (__bf16)val;
}

// W_j (V,N,N): B[kin][nout] = W_j[v][kin][nout]; two k-tiles of 32 per view.
__global__ void pack_wj_kernel(const float* __restrict__ Wj,
                               __bf16* __restrict__ pwj) {
  int i = blockIdx.x * blockDim.x + threadIdx.x;     // 16*4*2*512 = 65536
  int e  = i & 15;
  int l  = (i >> 4) & 31;
  int kt = (i >> 9) & 1;
  int ft = (i >> 10) & 3;
  int v  = i >> 12;
  int s = l >> 4, chunk = e >> 3, w = e & 7;
  int kin  = kt * 32 + chunk * 16 + s * 8 + w;       // 0..63
  int nout = ft * 16 + (l & 15);
  pwj[i] = (__bf16)Wj[((size_t)v * 64 + kin) * 64 + nout];
}

// x (B,T,V) fp32 -> xp (B,T,32) bf16, cols 16..31 zero (pad k-tile of inp).
__global__ void pack_x_kernel(const float* __restrict__ x,
                              __bf16* __restrict__ xp) {
  int i = blockIdx.x * blockDim.x + threadIdx.x;     // B*T*32
  int col = i & 31;
  int bt  = i >> 5;                                  // b*T + t
  float val = (col < VV) ? x[(size_t)bt * VV + col] : 0.f;
  xp[i] = (__bf16)val;
}

__global__ void init_state_kernel(__bf16* __restrict__ h0,
                                  float* __restrict__ c) {
  int i = blockIdx.x * blockDim.x + threadIdx.x;     // 256*1024
  h0[i] = (__bf16)0.0f;
  c[i]  = 0.0f;
}

// ---------------------------------------------------------------------------
// One recurrence step, fully fused:
//   grid = (v in [0,16), mb in [0,8)), block = 128 threads = 4 waves.
//   waves 0..2: gate GEMM (i/f/o), 32 batch rows x 64 feats, K=1056.
//               Two A row-tiles share each B fragment (2x B-reuse).
//   wave  3   : j GEMM (h[:,v,:] @ W_j[v]), 32 x 64, K=64.
//   then barrier + elementwise cell update: c (f32), h (bf16 x2).
// ---------------------------------------------------------------------------
__global__ __launch_bounds__(128)
void step_kernel(const __bf16* __restrict__ pw,     // packed gate weights
                 const __bf16* __restrict__ pwj,    // packed W_j
                 const __bf16* __restrict__ h_in,   // (B,1024) bf16
                 __bf16*       __restrict__ h_out,  // (B,1024) bf16
                 float*        __restrict__ c,      // (B,1024) f32
                 __bf16*       __restrict__ hs,     // (B,T,V,N) bf16 history
                 const __bf16* __restrict__ xp,     // (B,T,32) bf16 padded x
                 const float*  __restrict__ x,      // (B,T,V) f32
                 const float*  __restrict__ Uj,     // (V,1,N)
                 const float*  __restrict__ bj,     // (V,N)
                 const float*  __restrict__ Wib,
                 const float*  __restrict__ Wfb,
                 const float*  __restrict__ Wob,
                 int t) {
  __shared__ float preact[4][MTILE][64];            // i,f,o,j preactivations

  const int v    = blockIdx.x;
  const int mb   = blockIdx.y;
  const int tid  = threadIdx.x;
  const int wave = tid >> 5;
  const int lane = tid & 31;
  const int s    = lane >> 4;                        // half-wave side
  const int arow = lane & 15;                        // A-matrix row (batch)
  const int col  = lane & 15;                        // C-matrix column (feat)
  const int b0   = mb * MTILE + arow;                // row tile 0 batch row
  const int b1   = b0 + 16;                          // row tile 1 batch row

  if (wave < 3) {
    // ---- gate GEMM: 32x64 tile of gate `wave`, K loop over 33 tiles ----
    v8f a00 = {}, a01 = {}, a02 = {}, a03 = {};      // row tile 0
    v8f a10 = {}, a11 = {}, a12 = {}, a13 = {};      // row tile 1
    const __bf16* hrow0 = h_in + (size_t)b0 * 1024;
    const __bf16* hrow1 = h_in + (size_t)b1 * 1024;
    const __bf16* xr0   = xp + ((size_t)b0 * TT + t) * 32;
    const __bf16* xr1   = xp + ((size_t)b1 * TT + t) * 32;
    const __bf16* pwg   = pw + ((size_t)(wave * 64 + v * 4) * KT_GATE) * 512;
    for (int kt = 0; kt < KT_GATE; ++kt) {
      AFrag fa0, fa1;
      const __bf16* r0 = (kt < 32) ? (hrow0 + kt * 32) : xr0;
      const __bf16* r1 = (kt < 32) ? (hrow1 + kt * 32) : xr1;
      fa0.h[0] = *(const v8bf*)(r0 + s * 8);
      fa0.h[1] = *(const v8bf*)(r0 + 16 + s * 8);
      fa1.h[0] = *(const v8bf*)(r1 + s * 8);
      fa1.h[1] = *(const v8bf*)(r1 + 16 + s * 8);
      const __bf16* pb = pwg + (size_t)kt * 512 + lane * 16;
      v16bf w0 = *(const v16bf*)(pb);
      v16bf w1 = *(const v16bf*)(pb + 1 * KT_GATE * 512);
      v16bf w2 = *(const v16bf*)(pb + 2 * KT_GATE * 512);
      v16bf w3 = *(const v16bf*)(pb + 3 * KT_GATE * 512);
      a00 = __builtin_amdgcn_wmma_f32_16x16x32_bf16(false, fa0.v, false, w0, (short)0, a00, false, false);
      a10 = __builtin_amdgcn_wmma_f32_16x16x32_bf16(false, fa1.v, false, w0, (short)0, a10, false, false);
      a01 = __builtin_amdgcn_wmma_f32_16x16x32_bf16(false, fa0.v, false, w1, (short)0, a01, false, false);
      a11 = __builtin_amdgcn_wmma_f32_16x16x32_bf16(false, fa1.v, false, w1, (short)0, a11, false, false);
      a02 = __builtin_amdgcn_wmma_f32_16x16x32_bf16(false, fa0.v, false, w2, (short)0, a02, false, false);
      a12 = __builtin_amdgcn_wmma_f32_16x16x32_bf16(false, fa1.v, false, w2, (short)0, a12, false, false);
      a03 = __builtin_amdgcn_wmma_f32_16x16x32_bf16(false, fa0.v, false, w3, (short)0, a03, false, false);
      a13 = __builtin_amdgcn_wmma_f32_16x16x32_bf16(false, fa1.v, false, w3, (short)0, a13, false, false);
    }
    for (int r = 0; r < 8; ++r) {
      int m = r + 8 * s;                              // C layout: M=r(+8), N=lane
      preact[wave][m][ 0 + col]      = a00[r];
      preact[wave][m][16 + col]      = a01[r];
      preact[wave][m][32 + col]      = a02[r];
      preact[wave][m][48 + col]      = a03[r];
      preact[wave][m + 16][ 0 + col] = a10[r];
      preact[wave][m + 16][16 + col] = a11[r];
      preact[wave][m + 16][32 + col] = a12[r];
      preact[wave][m + 16][48 + col] = a13[r];
    }
  } else {
    // ---- j GEMM: h[:,v,:] (32x64) @ W_j[v] (64x64), K = 2 tiles ----
    v8f a00 = {}, a01 = {}, a02 = {}, a03 = {};
    v8f a10 = {}, a11 = {}, a12 = {}, a13 = {};
    const __bf16* hrow0 = h_in + (size_t)b0 * 1024 + v * 64;
    const __bf16* hrow1 = h_in + (size_t)b1 * 1024 + v * 64;
    const __bf16* pwb   = pwj + ((size_t)(v * 4) * 2) * 512;
    for (int kt = 0; kt < 2; ++kt) {
      AFrag fa0, fa1;
      int k0 = kt * 32 + s * 8;
      fa0.h[0] = *(const v8bf*)(hrow0 + k0);
      fa0.h[1] = *(const v8bf*)(hrow0 + k0 + 16);
      fa1.h[0] = *(const v8bf*)(hrow1 + k0);
      fa1.h[1] = *(const v8bf*)(hrow1 + k0 + 16);
      const __bf16* pb = pwb + (size_t)kt * 512 + lane * 16;
      v16bf w0 = *(const v16bf*)(pb);
      v16bf w1 = *(const v16bf*)(pb + 2 * 512);
      v16bf w2 = *(const v16bf*)(pb + 4 * 512);
      v16bf w3 = *(const v16bf*)(pb + 6 * 512);
      a00 = __builtin_amdgcn_wmma_f32_16x16x32_bf16(false, fa0.v, false, w0, (short)0, a00, false, false);
      a10 = __builtin_amdgcn_wmma_f32_16x16x32_bf16(false, fa1.v, false, w0, (short)0, a10, false, false);
      a01 = __builtin_amdgcn_wmma_f32_16x16x32_bf16(false, fa0.v, false, w1, (short)0, a01, false, false);
      a11 = __builtin_amdgcn_wmma_f32_16x16x32_bf16(false, fa1.v, false, w1, (short)0, a11, false, false);
      a02 = __builtin_amdgcn_wmma_f32_16x16x32_bf16(false, fa0.v, false, w2, (short)0, a02, false, false);
      a12 = __builtin_amdgcn_wmma_f32_16x16x32_bf16(false, fa1.v, false, w2, (short)0, a12, false, false);
      a03 = __builtin_amdgcn_wmma_f32_16x16x32_bf16(false, fa0.v, false, w3, (short)0, a03, false, false);
      a13 = __builtin_amdgcn_wmma_f32_16x16x32_bf16(false, fa1.v, false, w3, (short)0, a13, false, false);
    }
    for (int r = 0; r < 8; ++r) {
      int m = r + 8 * s;
      preact[3][m][ 0 + col]      = a00[r];
      preact[3][m][16 + col]      = a01[r];
      preact[3][m][32 + col]      = a02[r];
      preact[3][m][48 + col]      = a03[r];
      preact[3][m + 16][ 0 + col] = a10[r];
      preact[3][m + 16][16 + col] = a11[r];
      preact[3][m + 16][32 + col] = a12[r];
      preact[3][m + 16][48 + col] = a13[r];
    }
  }
  __syncthreads();

  // ---- elementwise cell update: 2048 elems (32 batch x 64 feat) ----
  for (int it = 0; it < 16; ++it) {
    int idx  = tid + it * 128;
    int f    = idx & 63;
    int brow = idx >> 6;                             // 0..31
    int bb   = mb * MTILE + brow;
    int gf   = v * 64 + f;
    float pi = preact[0][brow][f] + Wib[gf];
    float pf = preact[1][brow][f] + Wfb[gf];
    float po = preact[2][brow][f] + Wob[gf];
    float xv = x[((size_t)bb * TT + t) * VV + v];
    float pj = preact[3][brow][f] + xv * Uj[v * 64 + f] + bj[v * 64 + f];
    float jt  = tanhf(pj);
    float itg = 1.f / (1.f + expf(-pi));
    float ftg = 1.f / (1.f + expf(-pf));
    float otg = 1.f / (1.f + expf(-po));
    size_t ci = (size_t)bb * 1024 + gf;
    float cn = c[ci] * ftg + itg * jt;
    c[ci] = cn;
    float hn = otg * tanhf(cn);
    h_out[ci] = (__bf16)hn;
    hs[(((size_t)bb * TT + t) * VV + v) * NN + f] = (__bf16)hn;
  }
}

// ---------------------------------------------------------------------------
// Attention over T: per (b,v) block of 64 threads computes
// s[t] = exp(tanh(<hs[b,t,v,:], F_alpha_n[v]> + b)), normalizes over T,
// writes alphas and the pooled g_n[b,v,:].
// ---------------------------------------------------------------------------
__global__ __launch_bounds__(64)
void alpha_kernel(const __bf16* __restrict__ hs,
                  const float*  __restrict__ Fa,     // (V,N,1)
                  const float*  __restrict__ Fab,    // (V,1)
                  float*        __restrict__ out_alphas,  // (B,T,V)
                  float*        __restrict__ g_n) {       // (B,V,N)
  __shared__ float sc[TT];
  __shared__ float red[2];
  __shared__ float bcast;
  int bx = blockIdx.x;
  int b = bx >> 4, v = bx & 15;
  int tid = threadIdx.x;
  int wid = tid >> 5, lane = tid & 31;
  float wn   = Fa[v * 64 + tid];
  float bias = Fab[v];
  const __bf16* hbase = hs + (((size_t)b * TT) * VV + v) * NN;

  for (int t = 0; t < TT; ++t) {
    float p = (float)hbase[(size_t)t * VV * NN + tid] * wn;
    for (int off = 16; off > 0; off >>= 1) p += __shfl_xor(p, off, 32);
    if (lane == 0) red[wid] = p;
    __syncthreads();
    if (tid == 0) sc[t] = expf(tanhf(red[0] + red[1] + bias));
    __syncthreads();
  }
  float local = 0.f;
  for (int t = tid; t < TT; t += 64) local += sc[t];
  for (int off = 16; off > 0; off >>= 1) local += __shfl_xor(local, off, 32);
  if (lane == 0) red[wid] = local;
  __syncthreads();
  if (tid == 0) bcast = 1.f / (red[0] + red[1]);
  __syncthreads();
  float inv = bcast;
  for (int t = tid; t < TT; t += 64)
    out_alphas[((size_t)b * TT + t) * VV + v] = sc[t] * inv;
  float g = 0.f;
  for (int t = 0; t < TT; ++t)
    g += sc[t] * inv * (float)hbase[(size_t)t * VV * NN + tid];
  g_n[((size_t)b * VV + v) * NN + tid] = g;
}

// ---------------------------------------------------------------------------
// Head: per batch b, hg = [g_n, h_last] (V x 128); mu, betas, mean.
// ---------------------------------------------------------------------------
__global__ __launch_bounds__(128)
void head_kernel(const float*  __restrict__ g_n,
                 const __bf16* __restrict__ hs,
                 const float*  __restrict__ Phiw,   // (1,128)
                 const float*  __restrict__ Phib,   // (1,)
                 const float*  __restrict__ Fbw,    // (1,128)
                 const float*  __restrict__ Fbb,    // (1,)
                 float*        __restrict__ out_mean,   // (B,)
                 float*        __restrict__ out_betas)  // (B,V)
{
  __shared__ float mus[VV], bets[VV];
  __shared__ float red1[4], red2[4];
  __shared__ float binv;
  int b = blockIdx.x;
  int tid = threadIdx.x;
  int wid = tid >> 5, lane = tid & 31;
  float pwv = Phiw[tid], fwv = Fbw[tid];
  for (int v = 0; v < VV; ++v) {
    float hg;
    if (tid < 64) hg = g_n[((size_t)b * VV + v) * NN + tid];
    else          hg = (float)hs[(((size_t)b * TT + (TT - 1)) * VV + v) * NN + (tid - 64)];
    float p1 = hg * pwv, p2 = hg * fwv;
    for (int off = 16; off > 0; off >>= 1) {
      p1 += __shfl_xor(p1, off, 32);
      p2 += __shfl_xor(p2, off, 32);
    }
    if (lane == 0) { red1[wid] = p1; red2[wid] = p2; }
    __syncthreads();
    if (tid == 0) {
      float d1 = red1[0] + red1[1] + red1[2] + red1[3];
      float d2 = red2[0] + red2[1] + red2[2] + red2[3];
      mus[v]  = d1 + Phib[0];
      bets[v] = expf(tanhf(d2 + Fbb[0]));
    }
    __syncthreads();
  }
  if (tid == 0) {
    float bs = 0.f;
    for (int v2 = 0; v2 < VV; ++v2) bs += bets[v2];
    float inv = 1.f / bs;
    float m = 0.f;
    for (int v2 = 0; v2 < VV; ++v2) m += bets[v2] * inv * mus[v2];
    out_mean[b] = m;
    binv = inv;
  }
  __syncthreads();
  if (tid < VV) out_betas[(size_t)b * VV + tid] = bets[tid] * binv;
}

// ---------------------------------------------------------------------------
extern "C" void kernel_launch(void* const* d_in, const int* in_sizes, int n_in,
                              void* d_out, int out_size, void* d_ws, size_t ws_size,
                              hipStream_t stream) {
  const float* x    = (const float*)d_in[0];
  const float* U_j  = (const float*)d_in[1];
  const float* W_j  = (const float*)d_in[2];
  const float* b_j  = (const float*)d_in[3];
  const float* Wi_w = (const float*)d_in[4];
  const float* Wi_b = (const float*)d_in[5];
  const float* Wf_w = (const float*)d_in[6];
  const float* Wf_b = (const float*)d_in[7];
  const float* Wo_w = (const float*)d_in[8];
  const float* Wo_b = (const float*)d_in[9];
  const float* Fa   = (const float*)d_in[10];
  const float* Fab  = (const float*)d_in[11];
  const float* Fbw  = (const float*)d_in[12];
  const float* Fbb  = (const float*)d_in[13];
  const float* Phiw = (const float*)d_in[14];
  const float* Phib = (const float*)d_in[15];

  char* ws = (char*)d_ws;
  size_t off = 0;
  __bf16* pw  = (__bf16*)(ws + off); off += (size_t)3 * 64 * KT_GATE * 512 * sizeof(__bf16);
  __bf16* pwj = (__bf16*)(ws + off); off += (size_t)16 * 4 * 2 * 512 * sizeof(__bf16);
  __bf16* xp  = (__bf16*)(ws + off); off += (size_t)BB * TT * 32 * sizeof(__bf16);
  __bf16* h0  = (__bf16*)(ws + off); off += (size_t)BB * 1024 * sizeof(__bf16);
  __bf16* h1  = (__bf16*)(ws + off); off += (size_t)BB * 1024 * sizeof(__bf16);
  float*  cst = (float*)(ws + off);  off += (size_t)BB * 1024 * sizeof(float);
  float*  gnb = (float*)(ws + off);  off += (size_t)BB * VV * NN * sizeof(float);
  __bf16* hs  = (__bf16*)(ws + off); off += (size_t)BB * TT * VV * NN * sizeof(__bf16);
  (void)ws_size; (void)in_sizes; (void)n_in; (void)out_size;

  float* out_mean   = (float*)d_out;                     // (B,)
  float* out_alphas = out_mean + BB;                     // (B,T,V)
  float* out_betas  = out_alphas + (size_t)BB * TT * VV; // (B,V)

  pack_gates_kernel<<<(3 * 64 * KT_GATE * 512) / 256, 256, 0, stream>>>(Wi_w, Wf_w, Wo_w, pw);
  pack_wj_kernel<<<(16 * 4 * 2 * 512) / 256, 256, 0, stream>>>(W_j, pwj);
  pack_x_kernel<<<(BB * TT * 32) / 256, 256, 0, stream>>>(x, xp);
  init_state_kernel<<<(BB * 1024) / 256, 256, 0, stream>>>(h0, cst);

  for (int t = 0; t < TT; ++t) {
    const __bf16* hin = (t & 1) ? h1 : h0;
    __bf16* hout      = (t & 1) ? h0 : h1;
    step_kernel<<<dim3(VV, BB / MTILE), 128, 0, stream>>>(
        pw, pwj, hin, hout, cst, hs, xp, x, U_j, b_j, Wi_b, Wf_b, Wo_b, t);
  }

  alpha_kernel<<<BB * VV, 64, 0, stream>>>(hs, Fa, Fab, out_alphas, gnb);
  head_kernel<<<BB, 128, 0, stream>>>(gnb, hs, Phiw, Phib, Fbw, Fbb, out_mean, out_betas);
}
